// MultiHeadAttention_13340168421927
// MI455X (gfx1250) — matrix operand order repaired and checked by reading, not compiled, not run on hardware
//
#include <hip/hip_runtime.h>
#include <cstdint>

// ---------------------------------------------------------------------------
// CDNA5 (gfx1250, wave32) bf16 WMMA types & helpers
// ---------------------------------------------------------------------------
typedef __attribute__((ext_vector_type(16))) __bf16 v16bf;
typedef __attribute__((ext_vector_type(8)))  __bf16 v8bf;
typedef __attribute__((ext_vector_type(4)))  __bf16 v4bf;
typedef __attribute__((ext_vector_type(8)))  float  v8f;

static __device__ __forceinline__ v16bf cat16(v8bf lo, v8bf hi) {
  return __builtin_shufflevector(lo, hi, 0, 1, 2, 3, 4, 5, 6, 7,
                                          8, 9, 10, 11, 12, 13, 14, 15);
}
// Two 16-byte (8 x bf16) loads -> one 16-element WMMA fragment register set.
static __device__ __forceinline__ v16bf ldfrag(const __bf16* p0, const __bf16* p1) {
  return cat16(*(const v8bf*)p0, *(const v8bf*)p1);
}
static __device__ __forceinline__ v8f wmma_bf16(v16bf a, v16bf b, v8f c) {
  // D = A(16x32 bf16) * B(32x16 bf16) + C(16x16 f32)
  return __builtin_amdgcn_wmma_f32_16x16x32_bf16(
      /*neg_a=*/false, a, /*neg_b=*/false, b,
      /*c_mod=*/(short)0, c, /*reuse_a=*/false, /*reuse_b=*/false);
}

// ---------------------------------------------------------------------------
// CDNA5 async global->LDS copy (ASYNCcnt-tracked, no VGPR round-trip).
// Each enabled lane moves 16 bytes: LDS[ldsOff] = MEM[gaddr].
// ---------------------------------------------------------------------------
static __device__ __forceinline__ void async_ld_b128(unsigned ldsOff,
                                                     const void* gaddr) {
  asm volatile("global_load_async_to_lds_b128 %0, %1, off"
               :: "v"(ldsOff), "v"((unsigned long long)gaddr)
               : "memory");
}
static __device__ __forceinline__ void wait_async0() {
  asm volatile("s_wait_asynccnt 0x0" ::: "memory");
}

// ---------------------------------------------------------------------------
// One-shot f32 -> bf16 conversion (halves all GEMM operand traffic; keeps the
// GEMM staging as pure async 16B copies with no per-element converts).
// ---------------------------------------------------------------------------
__global__ __launch_bounds__(256) void cvt_f32_bf16(
    const float* __restrict__ in, __bf16* __restrict__ out, int n4) {
  const int i = blockIdx.x * 256 + threadIdx.x;
  if (i < n4) {
    const float4 v = ((const float4*)in)[i];
    v4bf o;
    o[0] = (__bf16)v.x; o[1] = (__bf16)v.y;
    o[2] = (__bf16)v.z; o[3] = (__bf16)v.w;
    ((v4bf*)out)[i] = o;
  }
}

// ---------------------------------------------------------------------------
// GEMM with bias:  C[M,N] = A[M,K] @ W[N,K]^T + bias[N]   (A, W bf16)
// Block: 256 threads (8 waves). Block tile 128x128, K step 32.
// Waves arranged 4x2; each wave computes a 32x64 tile = 2x4 WMMA accumulators
// (8 WMMAs per 12 ds_load_b128 per K-step).
// Double-buffered LDS fed by async global->LDS B128 copies: the K-step t+1
// tile streams in while the 8 WMMAs of step t execute.
// ---------------------------------------------------------------------------
template <typename OT>
__global__ __launch_bounds__(256) void gemm_bias_wmma(
    const __bf16* __restrict__ A, const __bf16* __restrict__ W,
    const float* __restrict__ bias, OT* __restrict__ C,
    int M, int N, int K) {
  constexpr int BM = 128, BN = 128, BK = 32, LDSD = 40;  // padded stride
  __shared__ __bf16 As[2][BM][LDSD];
  __shared__ __bf16 Bs[2][BN][LDSD];

  const int tid  = threadIdx.x;
  const int wv   = tid >> 5;        // wave 0..7
  const int lane = tid & 31;
  const int lm   = lane & 15;       // lane in 16-lane half
  const int hs   = lane >> 4;       // half select
  const int wr   = wv >> 1;         // wave row 0..3 (32 rows each)
  const int wc   = wv & 1;          // wave col 0..1 (64 cols each)
  const int bm   = blockIdx.x * BM;
  const int bn   = blockIdx.y * BN;

  // Per-thread staging coords: thread moves rows {r0, r0+64}, 8-col chunk c0.
  const int r0 = tid >> 2;          // 0..63
  const int c0 = (tid & 3) * 8;     // 0,8,16,24

  // Issue async copies of one K-step tile pair into LDS buffer `buf`.
  auto stage = [&](int buf, int kt) {
    const __bf16* ga = A + (size_t)(bm + r0) * K + kt + c0;
    const __bf16* gb = W + (size_t)(bn + r0) * K + kt + c0;
    async_ld_b128((unsigned)(uintptr_t)&As[buf][r0][c0],      ga);
    async_ld_b128((unsigned)(uintptr_t)&As[buf][r0 + 64][c0], ga + (size_t)64 * K);
    async_ld_b128((unsigned)(uintptr_t)&Bs[buf][r0][c0],      gb);
    async_ld_b128((unsigned)(uintptr_t)&Bs[buf][r0 + 64][c0], gb + (size_t)64 * K);
  };

  v8f acc[2][4];
#pragma unroll
  for (int mi = 0; mi < 2; ++mi)
#pragma unroll
    for (int ni = 0; ni < 4; ++ni)
#pragma unroll
      for (int r = 0; r < 8; ++r) acc[mi][ni][r] = 0.f;

  stage(0, 0);                       // prologue prefetch
  const int T = K / BK;
  for (int t = 0; t < T; ++t) {
    wait_async0();                   // this wave's async copies landed
    __syncthreads();                 // -> whole tile visible, prior reads done
    if (t + 1 < T) stage((t + 1) & 1, (t + 1) * BK);  // prefetch next step

    const int buf = t & 1;
    // A fragments: lane half 0 -> K{0..7,16..23}, half 1 -> K{8..15,24..31}
    v16bf af[2];
#pragma unroll
    for (int mi = 0; mi < 2; ++mi) {
      const int arow = wr * 32 + mi * 16 + lm;
      af[mi] = ldfrag(&As[buf][arow][hs * 8], &As[buf][arow][16 + hs * 8]);
    }
    // B fragments: lane = N col, half 0 -> K{0..7,16..23}, half 1 -> K{8..15,24..31}
    v16bf bfr[4];
#pragma unroll
    for (int ni = 0; ni < 4; ++ni) {
      const int bcol = wc * 64 + ni * 16 + lm;
      bfr[ni] = ldfrag(&Bs[buf][bcol][hs * 8], &Bs[buf][bcol][16 + hs * 8]);
    }
#pragma unroll
    for (int mi = 0; mi < 2; ++mi)
#pragma unroll
      for (int ni = 0; ni < 4; ++ni)
        acc[mi][ni] = wmma_bf16(af[mi], bfr[ni], acc[mi][ni]);
  }

  // Epilogue: C layout -> VGPR r holds row (r or r+8), col = lane&15
#pragma unroll
  for (int mi = 0; mi < 2; ++mi) {
    const int growb = bm + wr * 32 + mi * 16 + hs * 8;
#pragma unroll
    for (int ni = 0; ni < 4; ++ni) {
      const int gcol = bn + wc * 64 + ni * 16 + lm;
      const float bv = bias[gcol];
#pragma unroll
      for (int r = 0; r < 8; ++r) {
        const float v = acc[mi][ni][r] + bv;
        C[(size_t)(growb + r) * N + gcol] = (OT)v;
      }
    }
  }
}

// ---------------------------------------------------------------------------
// Flash attention (causal), bf16 WMMA, online softmax.
// Grid: (S/128, B*NH). Block: 256 threads = 8 waves; wave owns 16 Q rows.
// qkv: [B*S, 3H] bf16 (Q at col 0, K at H, V at 2H; head h -> cols h*128..)
// attn: [B*S, H] bf16
// ---------------------------------------------------------------------------
__global__ __launch_bounds__(256) void flash_attn_wmma(
    const __bf16* __restrict__ qkv, __bf16* __restrict__ attn) {
  constexpr int S = 2048, H = 2048, NH = 16, D = 128, QKVLD = 3 * H;
  constexpr float scale = 0.08838834764831845f;  // 1/sqrt(128)

  __shared__ __bf16 Kt[32][136];   // [key][d]   row-major K tile
  __shared__ __bf16 Vt[128][40];   // [d][key]   transposed V tile
  __shared__ __bf16 Ps[8][16][40]; // per-wave P scratch [row][key]

  const int tid  = threadIdx.x;
  const int wv   = tid >> 5;
  const int lane = tid & 31;
  const int lm   = lane & 15;
  const int hs   = lane >> 4;
  const int qb   = blockIdx.x * 128;       // Q tile base within S
  const int bh   = blockIdx.y;
  const int b    = bh / NH;
  const int h    = bh % NH;
  const size_t bS = (size_t)b * S;

  // Q fragments for the 4 K-steps over head_dim (kept in registers)
  const __bf16* qp = qkv + (bS + qb + wv * 16 + lm) * (size_t)QKVLD + h * D;
  v16bf qf[4];
#pragma unroll
  for (int s = 0; s < 4; ++s)
    qf[s] = ldfrag(qp + s * 32 + hs * 8, qp + s * 32 + 16 + hs * 8);

  v8f O[8];        // 16x128 f32 accumulator (8 d-tiles)
  float m[8], l[8];
#pragma unroll
  for (int d = 0; d < 8; ++d)
#pragma unroll
    for (int r = 0; r < 8; ++r) O[d][r] = 0.f;
#pragma unroll
  for (int r = 0; r < 8; ++r) { m[r] = -1e30f; l[r] = 0.f; }

  const int wqmax = qb + wv * 16 + 15;     // last Q row this wave owns
  const int njt   = (qb + 128) >> 5;       // causal: only key tiles <= tile end

  for (int j = 0; j < njt; ++j) {
    const int kb = j * 32;
    // ---- stage 32 keys: K row-major, V transposed ----
#pragma unroll
    for (int i = 0; i < 2; ++i) {
      const int g   = tid + i * 256;       // 0..511
      const int row = g >> 4;              // key 0..31
      const int cs  = (g & 15) * 8;        // d col 0..120
      const size_t gro = (bS + kb + row) * (size_t)QKVLD + h * D + cs;
      *(v8bf*)&Kt[row][cs] = *(const v8bf*)(qkv + gro + H);
      const v8bf vv = *(const v8bf*)(qkv + gro + 2 * H);
#pragma unroll
      for (int e = 0; e < 8; ++e) Vt[cs + e][row] = vv[e];
    }
    __syncthreads();

    if (kb <= wqmax) {  // wave-level causal skip
      // ---- scores: two 16-key subtiles, contract over d (4 WMMA steps) ----
      v8f s0, s1;
#pragma unroll
      for (int r = 0; r < 8; ++r) { s0[r] = 0.f; s1[r] = 0.f; }
#pragma unroll
      for (int s = 0; s < 4; ++s) {
        const v16bf b0 = ldfrag(&Kt[lm][s * 32 + hs * 8],
                                &Kt[lm][s * 32 + 16 + hs * 8]);
        s0 = wmma_bf16(qf[s], b0, s0);
        const v16bf b1 = ldfrag(&Kt[16 + lm][s * 32 + hs * 8],
                                &Kt[16 + lm][s * 32 + 16 + hs * 8]);
        s1 = wmma_bf16(qf[s], b1, s1);
      }

      // ---- joint online softmax over the full 32-key tile ----
      const int qrow0 = qb + wv * 16 + hs * 8;
#pragma unroll
      for (int r = 0; r < 8; ++r) {
        float a0 = s0[r] * scale, a1 = s1[r] * scale;
        if (kb + lm      > qrow0 + r) a0 = -3e30f;   // causal mask
        if (kb + 16 + lm > qrow0 + r) a1 = -3e30f;
        float mx = fmaxf(a0, a1);
        mx = fmaxf(mx, __shfl_xor(mx, 1, 32));
        mx = fmaxf(mx, __shfl_xor(mx, 2, 32));
        mx = fmaxf(mx, __shfl_xor(mx, 4, 32));
        mx = fmaxf(mx, __shfl_xor(mx, 8, 32));       // row max (16-lane half)
        const float mn   = fmaxf(m[r], mx);
        const float corr = __expf(m[r] - mn);
        m[r] = mn;
        const float p0 = __expf(a0 - mn), p1 = __expf(a1 - mn);
        float rs = p0 + p1;
        rs += __shfl_xor(rs, 1, 32);
        rs += __shfl_xor(rs, 2, 32);
        rs += __shfl_xor(rs, 4, 32);
        rs += __shfl_xor(rs, 8, 32);                 // row sum
        l[r] = l[r] * corr + rs;
#pragma unroll
        for (int d = 0; d < 8; ++d) O[d][r] *= corr; // rescale accumulator
        Ps[wv][hs * 8 + r][lm]      = (__bf16)p0;    // P -> LDS (A layout src)
        Ps[wv][hs * 8 + r][16 + lm] = (__bf16)p1;
      }

      // ---- O += P(16x32) @ V(32x128): 8 WMMAs ----
      const v16bf pf = ldfrag(&Ps[wv][lm][hs * 8], &Ps[wv][lm][16 + hs * 8]);
#pragma unroll
      for (int d = 0; d < 8; ++d) {
        const v16bf vf = ldfrag(&Vt[d * 16 + lm][hs * 8],
                                &Vt[d * 16 + lm][16 + hs * 8]);
        O[d] = wmma_bf16(pf, vf, O[d]);
      }
    }
    __syncthreads();
  }

  // ---- normalize and write attn output [B*S, H] bf16 ----
  const size_t orow0 = bS + qb + wv * 16 + hs * 8;
#pragma unroll
  for (int r = 0; r < 8; ++r) {
    const float inv = 1.0f / l[r];
    __bf16* op = attn + (orow0 + r) * (size_t)H + h * D + lm;
#pragma unroll
    for (int d = 0; d < 8; ++d) op[d * 16] = (__bf16)(O[d][r] * inv);
  }
}

// ---------------------------------------------------------------------------
// Launch: pre-convert to bf16 -> QKV proj -> flash attention -> out proj
// ---------------------------------------------------------------------------
extern "C" void kernel_launch(void* const* d_in, const int* in_sizes, int n_in,
                              void* d_out, int out_size, void* d_ws, size_t ws_size,
                              hipStream_t stream) {
  const float* hsin  = (const float*)d_in[0];  // [B,S,H] f32
  const float* W_in  = (const float*)d_in[1];  // [3H,H]  f32
  const float* b_in  = (const float*)d_in[2];  // [3H]
  const float* W_out = (const float*)d_in[3];  // [H,H]
  const float* b_out = (const float*)d_in[4];  // [H]
  float* out = (float*)d_out;                  // [B,S,H] f32

  constexpr int B = 2, S = 2048, H = 2048, M = B * S;
  constexpr size_t N_QKV  = (size_t)M * (3 * H);  // 25.2M
  constexpr size_t N_ATTN = (size_t)M * H;        //  8.4M
  constexpr size_t N_HID  = (size_t)M * H;        //  8.4M
  constexpr size_t N_WIN  = (size_t)(3 * H) * H;  // 12.6M
  constexpr size_t N_WOUT = (size_t)H * H;        //  4.2M

  char* ws = (char*)d_ws;
  __bf16* qkv  = (__bf16*)ws;                         ws += N_QKV  * 2;
  __bf16* attn = (__bf16*)ws;                         ws += N_ATTN * 2;
  __bf16* hbf  = (__bf16*)ws;                         ws += N_HID  * 2;
  __bf16* wibf = (__bf16*)ws;                         ws += N_WIN  * 2;
  __bf16* wobf = (__bf16*)ws;

  // One-shot operand downconversion (all sizes divisible by 1024)
  cvt_f32_bf16<<<(int)(N_HID  / 4 / 256), 256, 0, stream>>>(hsin,  hbf,  (int)(N_HID  / 4));
  cvt_f32_bf16<<<(int)(N_WIN  / 4 / 256), 256, 0, stream>>>(W_in,  wibf, (int)(N_WIN  / 4));
  cvt_f32_bf16<<<(int)(N_WOUT / 4 / 256), 256, 0, stream>>>(W_out, wobf, (int)(N_WOUT / 4));

  gemm_bias_wmma<__bf16>
      <<<dim3(M / 128, (3 * H) / 128), 256, 0, stream>>>(
          hbf, wibf, b_in, qkv, M, 3 * H, H);

  flash_attn_wmma<<<dim3(S / 128, B * 16), 256, 0, stream>>>(qkv, attn);

  gemm_bias_wmma<float>
      <<<dim3(M / 128, H / 128), 256, 0, stream>>>(
          attn, wobf, b_out, out, M, H, H);
}
